// QuantizedLinear_78134045048889
// MI455X (gfx1250) — 8-bit matrix operand order repaired and checked by reading, compiled, not run on hardware
//
#include <hip/hip_runtime.h>
#include <cstdint>

typedef __attribute__((ext_vector_type(8))) int   v8i;
typedef __attribute__((ext_vector_type(4))) float v4f;

#define QMAX 127.0f

#define BM 128
#define BN 128
#define BK 64
#define LDSS 80
#define BUFB (BM * LDSS)

#define WMMA_IU8(A, B, C) \
    __builtin_amdgcn_wmma_i32_16x16x64_iu8(true, (A), true, (B), (C), false, false)

__global__ void quantize_rows_kernel(const float* __restrict__ x,
                                     int8_t* __restrict__ xq,
                                     float* __restrict__ xscale,
                                     int Din) {
    const int t = blockIdx.x;
    const float* row = x + (size_t)t * Din;

    __shared__ float red[8];

    float amax = 0.0f;
    for (int i = threadIdx.x * 4; i < Din; i += blockDim.x * 4) {
        v4f v = __builtin_nontemporal_load((const v4f*)(row + i));
        amax = fmaxf(amax, fmaxf(fmaxf(fabsf(v.x), fabsf(v.y)),
                                 fmaxf(fabsf(v.z), fabsf(v.w))));
    }
    #pragma unroll
    for (int off = 16; off > 0; off >>= 1)
        amax = fmaxf(amax, __shfl_down(amax, off, 32));

    const int lane = threadIdx.x & 31;
    const int wv   = threadIdx.x >> 5;
    if (lane == 0) red[wv] = amax;
    __syncthreads();
    if (wv == 0) {
        float v = (lane < 8) ? red[lane] : 0.0f;
        #pragma unroll
        for (int off = 4; off > 0; off >>= 1)
            v = fmaxf(v, __shfl_down(v, off, 32));
        if (lane == 0) red[0] = v;
    }
    __syncthreads();

    const float scale = fmaxf(red[0], 1e-8f) / QMAX;
    const float inv   = 1.0f / scale;
    if (threadIdx.x == 0) xscale[t] = scale;

    int8_t* orow = xq + (size_t)t * Din;
    for (int i = threadIdx.x * 4; i < Din; i += blockDim.x * 4) {
        v4f v = *(const v4f*)(row + i);
        int a = (int)rintf(fminf(fmaxf(v.x * inv, -128.0f), 127.0f));
        int b = (int)rintf(fminf(fmaxf(v.y * inv, -128.0f), 127.0f));
        int c = (int)rintf(fminf(fmaxf(v.z * inv, -128.0f), 127.0f));
        int d = (int)rintf(fminf(fmaxf(v.w * inv, -128.0f), 127.0f));
        uint32_t packed = (uint32_t)(a & 0xFF) | ((uint32_t)(b & 0xFF) << 8) |
                          ((uint32_t)(c & 0xFF) << 16) | ((uint32_t)(d & 0xFF) << 24);
        *(uint32_t*)(orow + i) = packed;
    }
}

__device__ __forceinline__ v8i load_a_frag(const int8_t* p) {
    int4 t0 = *(const int4*)(p + 0);
    int4 t2 = *(const int4*)(p + 32);
    v8i a;
    a[0] = t0.x; a[1] = t0.y; a[2] = t0.z; a[3] = t0.w;
    a[4] = t2.x; a[5] = t2.y; a[6] = t2.z; a[7] = t2.w;
    return a;
}

__device__ __forceinline__ v8i load_b_frag(const int8_t* p) {
    int4 lo = *(const int4*)(p + 0);
    int4 hi = *(const int4*)(p + 32);
    v8i b;
    b[0] = lo.x; b[1] = lo.y; b[2] = lo.z; b[3] = lo.w;
    b[4] = hi.x; b[5] = hi.y; b[6] = hi.z; b[7] = hi.w;
    return b;
}

__global__ __launch_bounds__(256) void int8_gemm_lds_kernel(
    const int8_t* __restrict__ xq, const float* __restrict__ xscale,
    const int8_t* __restrict__ wq, const float* __restrict__ wscale,
    const float* __restrict__ bias, float* __restrict__ out,
    int T, int Din, int Dout) {
    __shared__ int8_t As[2 * BM * LDSS];
    __shared__ int8_t Bs[2 * BM * LDSS];

    const int tid  = threadIdx.x;
    const int wave = tid >> 5;
    const int lane = tid & 31;
    const int half = lane >> 4;
    const int l16  = lane & 15;

    const int tilesN = Dout / BN;
    const int blockM = (blockIdx.x / tilesN) * BM;
    const int blockN = (blockIdx.x % tilesN) * BN;

    const int wr = wave >> 1;
    const int wc = wave & 1;
    const int m0 = blockM + wr * 32;
    const int n0 = blockN + wc * 64;

    const int r0   = tid >> 2;
    const int koff = (tid & 3) * 16;
    const uint32_t gOff0 = (uint32_t)r0 * (uint32_t)Din + (uint32_t)koff;
    const uint32_t gOff1 = gOff0 + (uint32_t)64 * (uint32_t)Din;
    const int8_t* aG = xq + (size_t)blockM * Din;
    const int8_t* bG = wq + (size_t)blockN * Din;

    int8_t* aSt0 = As + r0 * LDSS + koff;
    int8_t* aSt1 = aSt0 + 64 * LDSS;
    int8_t* bSt0 = Bs + r0 * LDSS + koff;
    int8_t* bSt1 = bSt0 + 64 * LDSS;

    const int8_t* aRd = As + (wr * 32 + l16) * LDSS + half * 16;
    const int8_t* bRd = Bs + (wc * 64 + l16) * LDSS + half * 16;

    v8i acc00 = {}, acc01 = {}, acc02 = {}, acc03 = {};
    v8i acc10 = {}, acc11 = {}, acc12 = {}, acc13 = {};

    {
        int4 va0 = *(const int4*)(aG + gOff0);
        int4 va1 = *(const int4*)(aG + gOff1);
        int4 vb0 = *(const int4*)(bG + gOff0);
        int4 vb1 = *(const int4*)(bG + gOff1);
        *(int4*)(aSt0) = va0;  *(int4*)(aSt1) = va1;
        *(int4*)(bSt0) = vb0;  *(int4*)(bSt1) = vb1;
    }
    __syncthreads();

    const int nsteps = Din >> 6;

#define GEMM_STEP(BUF, KS)                                                     \
    {                                                                          \
           \
        const int _kn = ((KS) + 1 < nsteps) ? ((KS) + 1) : (KS);               \
        const int8_t* ag = aG + (_kn << 6);                                    \
        const int8_t* bg = bG + (_kn << 6);                                    \
        int4 va0 = *(const int4*)(ag + gOff0);                                 \
        int4 va1 = *(const int4*)(ag + gOff1);                                 \
        int4 vb0 = *(const int4*)(bg + gOff0);                                 \
        int4 vb1 = *(const int4*)(bg + gOff1);                                 \
                                  \
        const int _boff = (BUF) * BUFB;                                        \
        v8i a0 = load_a_frag(aRd + _boff);                                     \
        v8i a1 = load_a_frag(aRd + _boff + 16 * LDSS);                         \
        v8i b0 = load_b_frag(bRd + _boff);                                     \
        v8i b1 = load_b_frag(bRd + _boff + 16 * LDSS);                         \
        v8i b2 = load_b_frag(bRd + _boff + 32 * LDSS);                         \
        v8i b3 = load_b_frag(bRd + _boff + 48 * LDSS);                         \
                                        \
        const int _noff = ((BUF) ^ 1) * BUFB;                                  \
        *(int4*)(aSt0 + _noff) = va0;  *(int4*)(aSt1 + _noff) = va1;           \
        *(int4*)(bSt0 + _noff) = vb0;  *(int4*)(bSt1 + _noff) = vb1;           \
               \
        acc00 = WMMA_IU8(a0, b0, acc00);                                       \
        acc10 = WMMA_IU8(a1, b0, acc10);                                       \
        acc01 = WMMA_IU8(a0, b1, acc01);                                       \
        acc11 = WMMA_IU8(a1, b1, acc11);                                       \
        acc02 = WMMA_IU8(a0, b2, acc02);                                       \
        acc12 = WMMA_IU8(a1, b2, acc12);                                       \
        acc03 = WMMA_IU8(a0, b3, acc03);                                       \
        acc13 = WMMA_IU8(a1, b3, acc13);                                       \
        __syncthreads();                                                       \
    }

    for (int ks = 0; ks < nsteps; ks += 2) {
        GEMM_STEP(0, ks)
        GEMM_STEP(1, ks + 1)
    }
#undef GEMM_STEP

    v8i accs[2][4] = { { acc00, acc01, acc02, acc03 },
                       { acc10, acc11, acc12, acc13 } };
    #pragma unroll
    for (int i = 0; i < 2; ++i) {
        const int rbase = m0 + i * 16 + half * 8;
        const float* xsp = xscale + rbase;
        v4f s0 = *(const v4f*)(xsp + 0);
        v4f s1 = *(const v4f*)(xsp + 4);
        float rs[8] = { s0.x, s0.y, s0.z, s0.w, s1.x, s1.y, s1.z, s1.w };
        #pragma unroll
        for (int nt = 0; nt < 4; ++nt) {
            const int col = n0 + nt * 16 + l16;
            const float wsc = wscale[col];
            const float bv  = bias[col];
            #pragma unroll
            for (int r = 0; r < 8; ++r) {
                float v = fmaf((float)accs[i][nt][r] * rs[r], wsc, bv);
                __builtin_nontemporal_store(v, out + (size_t)(rbase + r) * Dout + col);
            }
        }
    }
}

__global__ void int8_gemm_reg_kernel(const int8_t* __restrict__ xq,
                                     const float* __restrict__ xscale,
                                     const int8_t* __restrict__ wq,
                                     const float* __restrict__ wscale,
                                     const float* __restrict__ bias,
                                     float* __restrict__ out,
                                     int T, int Din, int Dout) {
    const int wave  = threadIdx.x >> 5;
    const int lane  = threadIdx.x & 31;
    const int gwave = blockIdx.x * (blockDim.x >> 5) + wave;

    const int colGroups = Dout >> 6;
    const int m0 = (gwave / colGroups) * 32;
    const int n0 = (gwave % colGroups) * 64;
    if (m0 >= T) return;

    const int half = lane >> 4;
    const int l16  = lane & 15;

    const int8_t* ab0 = xq + (size_t)(m0 + l16) * Din + half * 16;
    const int8_t* ab1 = ab0 + (size_t)16 * Din;
    const int8_t* bb0 = wq + (size_t)(n0 + 0  + l16) * Din + half * 16;
    const int8_t* bb1 = wq + (size_t)(n0 + 16 + l16) * Din + half * 16;
    const int8_t* bb2 = wq + (size_t)(n0 + 32 + l16) * Din + half * 16;
    const int8_t* bb3 = wq + (size_t)(n0 + 48 + l16) * Din + half * 16;

    v8i acc00 = {}, acc01 = {}, acc02 = {}, acc03 = {};
    v8i acc10 = {}, acc11 = {}, acc12 = {}, acc13 = {};

    for (int k0 = 0; k0 < Din; k0 += 64) {
        v8i a0 = load_a_frag(ab0 + k0);
        v8i a1 = load_a_frag(ab1 + k0);
        v8i b0 = load_b_frag(bb0 + k0);
        v8i b1 = load_b_frag(bb1 + k0);
        v8i b2 = load_b_frag(bb2 + k0);
        v8i b3 = load_b_frag(bb3 + k0);
        acc00 = WMMA_IU8(a0, b0, acc00);
        acc10 = WMMA_IU8(a1, b0, acc10);
        acc01 = WMMA_IU8(a0, b1, acc01);
        acc11 = WMMA_IU8(a1, b1, acc11);
        acc02 = WMMA_IU8(a0, b2, acc02);
        acc12 = WMMA_IU8(a1, b2, acc12);
        acc03 = WMMA_IU8(a0, b3, acc03);
        acc13 = WMMA_IU8(a1, b3, acc13);
    }

    v8i accs[2][4] = { { acc00, acc01, acc02, acc03 },
                       { acc10, acc11, acc12, acc13 } };
    #pragma unroll
    for (int i = 0; i < 2; ++i) {
        const int rbase = m0 + i * 16 + half * 8;
        const float* xsp = xscale + rbase;
        v4f s0 = *(const v4f*)(xsp + 0);
        v4f s1 = *(const v4f*)(xsp + 4);
        float rs[8] = { s0.x, s0.y, s0.z, s0.w, s1.x, s1.y, s1.z, s1.w };
        #pragma unroll
        for (int nt = 0; nt < 4; ++nt) {
            const int col = n0 + nt * 16 + l16;
            const float wsc = wscale[col];
            const float bv  = bias[col];
            #pragma unroll
            for (int r = 0; r < 8; ++r) {
                float v = fmaf((float)accs[i][nt][r] * rs[r], wsc, bv);
                __builtin_nontemporal_store(v, out + (size_t)(rbase + r) * Dout + col);
            }
        }
    }
}

extern "C" void kernel_launch(void* const* d_in, const int* in_sizes, int n_in,
                              void* d_out, int out_size, void* d_ws, size_t ws_size,
                              hipStream_t stream) {
    const float*  x      = (const float*)d_in[0];
    const int8_t* wq     = (const int8_t*)d_in[1];
    const float*  wscale = (const float*)d_in[2];
    const float*  bias   = (const float*)d_in[3];
    float*        out    = (float*)d_out;

    const int Dout = in_sizes[2];
    const int Din  = in_sizes[1] / Dout;
    const int T    = in_sizes[0] / Din;

    float* xscale = (float*)d_ws;
    size_t xq_off = ((size_t)T * sizeof(float) + 255) & ~(size_t)255;
    int8_t* xq    = (int8_t*)d_ws + xq_off;

    quantize_rows_kernel<<<T, 256, 0, stream>>>(x, xq, xscale, Din);

    if ((T % BM) == 0 && (Dout % BN) == 0 && (Din % 128) == 0) {
        const int blocks = (T / BM) * (Dout / BN);
        int8_gemm_lds_kernel<<<blocks, 256, 0, stream>>>(
            xq, xscale, wq, wscale, bias, out, T, Din, Dout);
    } else {
        const int totalWaves = ((T + 31) / 32) * (Dout / 64);
        const int blocks = (totalWaves + 3) / 4;
        int8_gemm_reg_kernel<<<blocks, 128, 0, stream>>>(
            xq, xscale, wq, wscale, bias, out, T, Din, Dout);
    }
}
